// MoE_74105365725748
// MI455X (gfx1250) — compile-verified, matrix operand order checked by reading, not run on hardware
//
#include <hip/hip_runtime.h>
#include <hip/hip_fp16.h>

typedef _Float16 h16;
typedef __attribute__((ext_vector_type(16))) _Float16 v16h;
typedef __attribute__((ext_vector_type(8)))  _Float16 v8h;
typedef __attribute__((ext_vector_type(2)))  _Float16 v2h;
typedef __attribute__((ext_vector_type(8)))  float    v8f;

#define NB 2048      // batch
#define NE 16        // experts
#define NCLS 10      // classes
#define TCH 256      // trunk channels
#define CAPX 256
#define MINTOK 102
#define LDSTRIDE 56  // halves; 112B row stride: 16B-aligned & bank-conflict-free

// Build a v16h fragment from two aligned 16B LDS chunks (2x ds_load_b128).
__device__ __forceinline__ v16h frag_ld(const h16* p /*row base + kb*/)
{
  v8h lo = *(const v8h*)(p);
  v8h hi = *(const v8h*)(p + 16);
  return __builtin_shufflevector(lo, hi, 0,1,2,3,4,5,6,7,8,9,10,11,12,13,14,15);
}

// Async 16B global -> LDS copy (CDNA5 GLOBAL_LOAD_ASYNC_TO_LDS_B128, ASYNCcnt).
__device__ __forceinline__ void async_b128(h16* lds_dst, const h16* gsrc)
{
  unsigned lds_addr = (unsigned)(uintptr_t)lds_dst;   // low 32 bits = LDS offset
  asm volatile("global_load_async_to_lds_b128 %0, %1, off"
               :: "v"(lds_addr), "v"(gsrc) : "memory");
}
__device__ __forceinline__ void async_wait0()
{
  asm volatile("s_wait_asynccnt 0x0" ::: "memory");
}

// ---------------------------------------------------------------------------
// Weight prep: f32 [COUT][K] -> f16 [COUT][KP] (K padded to multiple of 32)
// ---------------------------------------------------------------------------
__global__ __launch_bounds__(256)
void prep_weights_kernel(const float* __restrict__ w, h16* __restrict__ wh,
                         int K, int KP, int total)
{
  int i = blockIdx.x * blockDim.x + threadIdx.x;
  if (i >= total) return;
  int n = i / KP, kp = i % KP;
  wh[i] = (kp < K) ? (h16)w[(size_t)n * K + kp] : (h16)0.f;
}

// ---------------------------------------------------------------------------
// Implicit-GEMM conv + BN + ReLU, f16 in (or f32 for layer 1), f16 out.
// Block: 256 thr = 8 waves. Tile: M=128 pixels x NTILE<=64 channels.
// A staged row-major [row][k] (im2col gather), B tile async-DMA'd from the
// pre-converted f16 weights [n][kp]; both LDS stride 56 halves.
// ---------------------------------------------------------------------------
template<int CIN, int COUT, int HDIM, bool IN_F32>
__global__ __launch_bounds__(256)
void conv_bn_relu_wmma(const void* __restrict__ inp,
                       const h16* __restrict__ wh,
                       const float* __restrict__ cb, const float* __restrict__ g,
                       const float* __restrict__ be, const float* __restrict__ mm,
                       const float* __restrict__ vv,
                       h16* __restrict__ outp)
{
  constexpr int H = HDIM, W = HDIM, HWS = H * W;
  constexpr int K = CIN * 9;
  constexpr int KSTEPS = (K + 31) / 32;
  constexpr int KP = KSTEPS * 32;
  constexpr int NTILE = (COUT < 64) ? COUT : 64;
  constexpr int NFRAG = NTILE / 16;

  __shared__ __align__(16) h16 As[128 * LDSTRIDE];
  __shared__ __align__(16) h16 Bs[64 * LDSTRIDE];

  const int tid  = threadIdx.x;
  const int wave = tid >> 5;
  const int lane = tid & 31;
  const int m0 = blockIdx.x * 128;
  const int n0 = blockIdx.y * NTILE;

  v8f acc[NFRAG];
  const v8f vzero = {0.f,0.f,0.f,0.f,0.f,0.f,0.f,0.f};
  for (int f = 0; f < NFRAG; ++f) acc[f] = vzero;

  const int mrow = wave * 16 + (lane & 15);
  const int kb   = (lane < 16) ? 0 : 8;

  for (int ks = 0; ks < KSTEPS; ++ks) {
    // ---- B tile: async DMA, 16B per lane, NTILE*4 chunks ----
    if (tid < NTILE * 4) {
      int n = tid >> 2, q = (tid & 3) * 8;
      async_b128(Bs + n * LDSTRIDE + q,
                 wh + (size_t)(n0 + n) * KP + ks * 32 + q);
    }
    // ---- A tile (im2col gather): 128x32 ----
#pragma unroll
    for (int e = 0; e < 16; ++e) {
      int idx = tid + e * 256;
      int r = idx >> 5, kk = idx & 31;
      int kg = ks * 32 + kk;
      float val = 0.f;
      if (kg < K) {
        int ci = kg / 9, rem = kg % 9;
        int ky = rem / 3, kx = rem % 3;
        int m = m0 + r;
        int img = m / HWS, p = m % HWS;
        int py = p / W, px = p % W;
        int iy = py + ky - 1, ix = px + kx - 1;
        if (iy >= 0 && iy < H && ix >= 0 && ix < W) {
          size_t a = (size_t)(img * CIN + ci) * HWS + (size_t)iy * W + ix;
          val = IN_F32 ? ((const float*)inp)[a] : (float)((const h16*)inp)[a];
        }
      }
      As[r * LDSTRIDE + kk] = (h16)val;
    }
    // prefetch next K-step weight tile into cache
    if (ks + 1 < KSTEPS) {
      int n = tid & (NTILE - 1);
      __builtin_prefetch(wh + (size_t)(n0 + n) * KP + (ks + 1) * 32, 0, 1);
    }
    async_wait0();
    __syncthreads();

    const v16h a = frag_ld(As + mrow * LDSTRIDE + kb);
#pragma unroll
    for (int f = 0; f < NFRAG; ++f) {
      const int ncol = f * 16 + (lane & 15);
      const v16h b = frag_ld(Bs + ncol * LDSTRIDE + kb);
      acc[f] = __builtin_amdgcn_wmma_f32_16x16x32_f16(
                 false, a, false, b, (short)0, acc[f], false, false);
    }
    __syncthreads();
  }

  // ---- fused bias + eval-mode BN + ReLU epilogue (16B vector stores) ----
#pragma unroll
  for (int f = 0; f < NFRAG; ++f) {
    const int cout = n0 + f * 16 + (lane & 15);
    const float sc = g[cout] * rsqrtf(vv[cout] + 1e-5f);
    const float sh = (cb[cout] - mm[cout]) * sc + be[cout];
    const int mbase = m0 + wave * 16 + ((lane >= 16) ? 8 : 0);
    const int img = mbase / HWS, p = mbase % HWS;   // 8 consecutive pixels
    v8h ov;
#pragma unroll
    for (int vg = 0; vg < 8; ++vg) {
      float y = acc[f][vg] * sc + sh;
      ov[vg] = (h16)((y > 0.f) ? y : 0.f);
    }
    *(v8h*)(outp + (size_t)(img * COUT + cout) * HWS + p) = ov;
  }
}

// ---------------------------------------------------------------------------
// 2x2 max pool, f16 -> f16, NCHW (pixel pairs loaded as 4B vectors)
// ---------------------------------------------------------------------------
template<int C, int H>
__global__ __launch_bounds__(256)
void maxpool2_kernel(const h16* __restrict__ in, h16* __restrict__ out)
{
  constexpr int HO = H / 2;
  size_t i = (size_t)blockIdx.x * blockDim.x + threadIdx.x;
  size_t total = (size_t)NB * C * HO * HO;
  if (i >= total) return;
  int px = (int)(i % HO);
  size_t t = i / HO;
  int py = (int)(t % HO);
  t /= HO;                               // t = b*C + c
  const h16* base = in + t * (size_t)(H * H);
  int y0 = py * 2, x0 = px * 2;
  v2h top = *(const v2h*)(base + y0 * H + x0);
  v2h bot = *(const v2h*)(base + (y0 + 1) * H + x0);
  float m = fmaxf(fmaxf((float)top[0], (float)top[1]),
                  fmaxf((float)bot[0], (float)bot[1]));
  out[i] = (h16)m;
}

// ---------------------------------------------------------------------------
// Global average pool 8x8 -> feats (f32 + f16 copies), vector loads
// ---------------------------------------------------------------------------
__global__ __launch_bounds__(256)
void gap_kernel(const h16* __restrict__ in, float* __restrict__ feats,
                h16* __restrict__ featsh)
{
  int i = blockIdx.x * blockDim.x + threadIdx.x;    // b*TCH + c
  const v8h* base = (const v8h*)(in + (size_t)i * 64);
  float s = 0.f;
#pragma unroll
  for (int q = 0; q < 8; ++q) {
    v8h v = base[q];
#pragma unroll
    for (int e = 0; e < 8; ++e) s += (float)v[e];
  }
  s *= (1.f / 64.f);
  feats[i]  = s;
  featsh[i] = (h16)s;
}

// ---------------------------------------------------------------------------
// Head GEMM: Y[2048,N] = X[2048,256]f16 * W^T, optional bias/ReLU.
// MODE 0: W[n][k] = Wt[n*256+k]            (cls heads, N=160)
// MODE 1: W[n][k] = gw1[(n>>7)*32768 + k*128 + (n&127)]  (gate hidden, N=2048)
// A tile async-DMA'd (contiguous f16), B tile converted f32->f16 in VALU.
// ---------------------------------------------------------------------------
template<int NDIM, int MODE, bool RELU>
__global__ __launch_bounds__(256)
void head_gemm_wmma(const h16* __restrict__ X, const float* __restrict__ Wt,
                    const float* __restrict__ bias, float* __restrict__ Y)
{
  __shared__ __align__(16) h16 As[128 * LDSTRIDE];
  __shared__ __align__(16) h16 Bs[64 * LDSTRIDE];
  const int tid  = threadIdx.x;
  const int wave = tid >> 5;
  const int lane = tid & 31;
  const int m0 = blockIdx.x * 128;
  const int n0 = blockIdx.y * 64;

  v8f acc[4];
  const v8f vzero = {0.f,0.f,0.f,0.f,0.f,0.f,0.f,0.f};
  for (int f = 0; f < 4; ++f) acc[f] = vzero;

  const int mrow = wave * 16 + (lane & 15);
  const int kb   = (lane < 16) ? 0 : 8;

  for (int ks = 0; ks < 8; ++ks) {      // K = 256
    // A tile: 128 rows x 32 halves via async 16B chunks (512 chunks)
#pragma unroll
    for (int e = 0; e < 2; ++e) {
      int idx = tid + e * 256;          // 0..511
      int r = idx >> 2, kq = (idx & 3) * 8;
      async_b128(As + r * LDSTRIDE + kq,
                 X + (size_t)(m0 + r) * TCH + ks * 32 + kq);
    }
    // B tile transposed to [n][k] (needs f32->f16 conversion)
    for (int idx = tid; idx < 64 * 32; idx += 256) {
      int n = idx >> 5, kk = idx & 31;
      int ng = n0 + n, kg = ks * 32 + kk;
      float val = 0.f;
      if (ng < NDIM) {
        if (MODE == 0) val = Wt[(size_t)ng * TCH + kg];
        else           val = Wt[(size_t)(ng >> 7) * 32768 + (size_t)kg * 128 + (ng & 127)];
      }
      Bs[n * LDSTRIDE + kk] = (h16)val;
    }
    async_wait0();
    __syncthreads();

    const v16h a = frag_ld(As + mrow * LDSTRIDE + kb);
#pragma unroll
    for (int f = 0; f < 4; ++f) {
      const int ncol = f * 16 + (lane & 15);
      const v16h b = frag_ld(Bs + ncol * LDSTRIDE + kb);
      acc[f] = __builtin_amdgcn_wmma_f32_16x16x32_f16(
                 false, a, false, b, (short)0, acc[f], false, false);
    }
    __syncthreads();
  }

#pragma unroll
  for (int f = 0; f < 4; ++f) {
    const int n = n0 + f * 16 + (lane & 15);
    if (n < NDIM) {
      const float bv = bias[n];
      const int mbase = m0 + wave * 16 + ((lane >= 16) ? 8 : 0);
#pragma unroll
      for (int vg = 0; vg < 8; ++vg) {
        float y = acc[f][vg] + bv;
        if (RELU) y = (y > 0.f) ? y : 0.f;
        Y[(size_t)(mbase + vg) * NDIM + n] = y;
      }
    }
  }
}

// ---------------------------------------------------------------------------
// Per-(token,expert) softmax entropy -> confidence = -entropy
// ---------------------------------------------------------------------------
__global__ __launch_bounds__(256)
void entropy_kernel(const float* __restrict__ logits_e, float* __restrict__ conf)
{
  int i = blockIdx.x * blockDim.x + threadIdx.x;   // b*NE + e
  int b = i >> 4, e = i & 15;
  const float* l = logits_e + (size_t)b * (NE * NCLS) + e * NCLS;
  float mx = -3.0e38f;
#pragma unroll
  for (int c = 0; c < NCLS; ++c) mx = fmaxf(mx, l[c]);
  float p[NCLS]; float s = 0.f;
#pragma unroll
  for (int c = 0; c < NCLS; ++c) { p[c] = __expf(l[c] - mx); s += p[c]; }
  float inv = 1.f / s, ent = 0.f;
#pragma unroll
  for (int c = 0; c < NCLS; ++c) {
    float pc = p[c] * inv;
    ent -= pc * __logf(fmaxf(pc, 1e-12f));
  }
  conf[i] = -ent;
}

// ---------------------------------------------------------------------------
// Gate score + routing score fusion
// ---------------------------------------------------------------------------
__global__ __launch_bounds__(256)
void scores_kernel(const float* __restrict__ hidden, const float* __restrict__ gw2,
                   const float* __restrict__ gb2, const float* __restrict__ conf,
                   const float* __restrict__ usage, float* __restrict__ routing)
{
  int i = blockIdx.x * blockDim.x + threadIdx.x;   // b*NE + e
  int b = i >> 4, e = i & 15;
  const float* h = hidden + (size_t)b * (NE * 128) + e * 128;
  const float* w = gw2 + e * 128;
  float s = 0.f;
  for (int k = 0; k < 128; ++k) s += h[k] * w[k];
  float es = (s + gb2[e]) * 0.5f;                  // / DTEMP
  routing[i] = 0.6f * es + 0.4f * conf[i] - 2.0f * usage[e];
}

// ---------------------------------------------------------------------------
// Routing: single workgroup, 1024 threads (32 waves), radix-select top-k.
// ---------------------------------------------------------------------------
__device__ __forceinline__ unsigned okey(float f)
{
  unsigned u = __float_as_uint(f);
  return (u & 0x80000000u) ? ~u : (u | 0x80000000u);
}

__device__ __forceinline__ int block_reduce_int(int v, int* red, int tid)
{
  for (int off = 16; off; off >>= 1) v += __shfl_down(v, off, 32);
  if ((tid & 31) == 0) red[tid >> 5] = v;
  __syncthreads();
  if ((tid >> 5) == 0) {
    int x = red[tid];
    for (int off = 16; off; off >>= 1) x += __shfl_down(x, off, 32);
    if (tid == 0) red[0] = x;
  }
  __syncthreads();
  int r = red[0];
  __syncthreads();
  return r;
}

__global__ __launch_bounds__(1024)
void route_kernel(const float* __restrict__ routing, float* __restrict__ outD)
{
  __shared__ unsigned int Dm[NB];
  __shared__ unsigned char avail[NB];
  __shared__ float loads[NE];
  __shared__ float loads_snap[NE];
  __shared__ int red[32];
  __shared__ int tiecnt;

  const int tid = threadIdx.x;
  const int t0 = tid, t1 = tid + 1024;
  Dm[t0] = 0u; Dm[t1] = 0u;
  avail[t0] = 1; avail[t1] = 1;
  if (tid < NE) loads[tid] = 0.f;
  __syncthreads();

  for (int it = 0; it < 3; ++it) {
    if (tid < NE) loads_snap[tid] = loads[tid];
    __syncthreads();
    for (int j = 0; j < NE; ++j) {
      const float pen = (float)(it + 1) * 2.0f * loads_snap[j];
      const unsigned k0 = avail[t0] ? okey(routing[t0 * NE + j] - pen) : 0u;
      const unsigned k1 = avail[t1] ? okey(routing[t1 * NE + j] - pen) : 0u;

      const int vc = block_reduce_int((int)avail[t0] + (int)avail[t1], red, tid);
      int rc = CAPX - (int)loads[j];
      if (rc < 0) rc = 0;
      int ntc = 0;
      if (rc != 0 && vc != 0) {
        int a = (rc < vc) ? rc : vc;
        int b = (MINTOK < vc) ? MINTOK : vc;
        ntc = (a > b) ? a : b;
      }
      if (ntc > 0) {
        unsigned cur = 0u;
        for (int bit = 31; bit >= 0; --bit) {
          const unsigned cand = cur | (1u << bit);
          const int cnt = block_reduce_int((int)(k0 >= cand) + (int)(k1 >= cand), red, tid);
          if (cnt >= ntc) cur = cand;
        }
        const int cg = block_reduce_int((int)(k0 > cur) + (int)(k1 > cur), red, tid);
        const int tie_slots = ntc - cg;
        if (tid == 0) tiecnt = 0;
        __syncthreads();
        bool take0 = (k0 > cur);
        if (!take0 && avail[t0] && k0 == cur) {
          int slot = atomicAdd(&tiecnt, 1);
          take0 = (slot < tie_slots);
        }
        if (take0) { Dm[t0] |= (1u << j); avail[t0] = 0; }
        bool take1 = (k1 > cur);
        if (!take1 && avail[t1] && k1 == cur) {
          int slot = atomicAdd(&tiecnt, 1);
          take1 = (slot < tie_slots);
        }
        if (take1) { Dm[t1] |= (1u << j); avail[t1] = 0; }
        __syncthreads();
        if (tid == 0) loads[j] += (float)ntc;
        __syncthreads();
      }
    }
  }
  __syncthreads();

  // sequential fallback: least-loaded of top-3 (thread 0)
  if (tid == 0) {
    for (int i = 0; i < NB; ++i) {
      if (!avail[i]) continue;
      float v1 = -3.0e38f, v2 = -3.0e38f, v3 = -3.0e38f;
      int i1 = 0, i2 = 0, i3 = 0;
      for (int j = 0; j < NE; ++j) {
        float s = routing[i * NE + j];
        if (s > v1)      { v3 = v2; i3 = i2; v2 = v1; i2 = i1; v1 = s; i1 = j; }
        else if (s > v2) { v3 = v2; i3 = i2; v2 = s; i2 = j; }
        else if (s > v3) { v3 = s; i3 = j; }
      }
      int best = i1; float bl = loads[i1];
      if (loads[i2] < bl) { best = i2; bl = loads[i2]; }
      if (loads[i3] < bl) { best = i3; }
      Dm[i] |= (1u << best);
      loads[best] += 1.f;
    }
  }
  __syncthreads();

  for (int j = 0; j < NE; ++j) {
    outD[t0 * NE + j] = ((Dm[t0] >> j) & 1u) ? 1.f : 0.f;
    outD[t1 * NE + j] = ((Dm[t1] >> j) & 1u) ? 1.f : 0.f;
  }
}

// ---------------------------------------------------------------------------
// Final logits: masked softmax over assigned experts, blend class logits
// ---------------------------------------------------------------------------
__global__ __launch_bounds__(256)
void final_kernel(const float* __restrict__ routing, const float* __restrict__ Darr,
                  const float* __restrict__ logits_e, float* __restrict__ out)
{
  int b = blockIdx.x * blockDim.x + threadIdx.x;
  if (b >= NB) return;
  float a[NE], d[NE];
  float mx = -3.0e38f;
#pragma unroll
  for (int e = 0; e < NE; ++e) {
    d[e] = Darr[b * NE + e];
    a[e] = routing[b * NE + e] * d[e];
    mx = fmaxf(mx, a[e]);
  }
  float w[NE]; float s = 0.f;
#pragma unroll
  for (int e = 0; e < NE; ++e) {
    w[e] = __expf(a[e] - mx + (d[e] - 1.f) * 1e9f);
    s += w[e];
  }
  const float inv = 1.f / s;
#pragma unroll
  for (int c = 0; c < NCLS; ++c) {
    float acc = 0.f;
#pragma unroll
    for (int e = 0; e < NE; ++e)
      acc += w[e] * logits_e[(size_t)b * (NE * NCLS) + e * NCLS + c];
    out[b * NCLS + c] = acc * inv;
  }
}

// ---------------------------------------------------------------------------
extern "C" void kernel_launch(void* const* d_in, const int* in_sizes, int n_in,
                              void* d_out, int out_size, void* d_ws, size_t ws_size,
                              hipStream_t stream)
{
  const float* x    = (const float*)d_in[0];
  const float* cw1  = (const float*)d_in[1];
  const float* cb1  = (const float*)d_in[2];
  const float* g1   = (const float*)d_in[3];
  const float* be1  = (const float*)d_in[4];
  const float* m1   = (const float*)d_in[5];
  const float* v1   = (const float*)d_in[6];
  const float* cw2  = (const float*)d_in[7];
  const float* cb2  = (const float*)d_in[8];
  const float* g2   = (const float*)d_in[9];
  const float* be2  = (const float*)d_in[10];
  const float* m2   = (const float*)d_in[11];
  const float* v2   = (const float*)d_in[12];
  const float* cw3  = (const float*)d_in[13];
  const float* cb3  = (const float*)d_in[14];
  const float* g3   = (const float*)d_in[15];
  const float* be3  = (const float*)d_in[16];
  const float* m3   = (const float*)d_in[17];
  const float* v3   = (const float*)d_in[18];
  const float* cw4  = (const float*)d_in[19];
  const float* cb4  = (const float*)d_in[20];
  const float* g4   = (const float*)d_in[21];
  const float* be4  = (const float*)d_in[22];
  const float* m4   = (const float*)d_in[23];
  const float* v4   = (const float*)d_in[24];
  const float* gw1  = (const float*)d_in[25];
  const float* gb1  = (const float*)d_in[26];
  const float* gw2  = (const float*)d_in[27];
  const float* gb2  = (const float*)d_in[28];
  const float* clsw = (const float*)d_in[29];
  const float* clsb = (const float*)d_in[30];
  const float* usage= (const float*)d_in[31];

  char* ws = (char*)d_ws;
  const size_t RSZ = (size_t)268435456;          // 256 MB ping-pong regions
  h16* R0 = (h16*)ws;
  h16* R1 = (h16*)(ws + RSZ);

  // tail region: pre-converted f16 weights + small post-trunk buffers
  char* TB = ws + 2 * RSZ;
  h16* wh1 = (h16*)TB;                           //  32 x   32 =   1024
  h16* wh2 = wh1 + 1024;                         //  64 x  288 =  18432
  h16* wh3 = wh2 + 18432;                        // 128 x  576 =  73728
  h16* wh4 = wh3 + 73728;                        // 256 x 1152 = 294912
  float* feats    = (float*)(TB + ((size_t)1 << 20));
  h16*   featsh   = (h16*)  (TB + ((size_t)3 << 20));
  float* logits_e = (float*)(TB + ((size_t)4 << 20));
  float* conf     = (float*)(TB + ((size_t)6 << 20));
  float* hidden   = (float*)(TB + ((size_t)7 << 20));

  float* out      = (float*)d_out;
  float* routing  = out + NB * NCLS;             // 20480
  float* Dout     = routing + NB * NE;           // +32768

  // weight pre-conversion (tiny)
  prep_weights_kernel<<<(32 * 32 + 255) / 256, 256, 0, stream>>>(cw1, wh1, 27, 32, 32 * 32);
  prep_weights_kernel<<<(64 * 288 + 255) / 256, 256, 0, stream>>>(cw2, wh2, 288, 288, 64 * 288);
  prep_weights_kernel<<<(128 * 576 + 255) / 256, 256, 0, stream>>>(cw3, wh3, 576, 576, 128 * 576);
  prep_weights_kernel<<<(256 * 1152 + 255) / 256, 256, 0, stream>>>(cw4, wh4, 1152, 1152, 256 * 1152);

  // trunk
  conv_bn_relu_wmma<3, 32, 32, true ><<<dim3(16384, 1), 256, 0, stream>>>(x,  wh1, cb1, g1, be1, m1, v1, R0);
  conv_bn_relu_wmma<32, 64, 32, false><<<dim3(16384, 1), 256, 0, stream>>>(R0, wh2, cb2, g2, be2, m2, v2, R1);
  maxpool2_kernel<64, 32><<<131072, 256, 0, stream>>>(R1, R0);
  conv_bn_relu_wmma<64, 128, 16, false><<<dim3(4096, 2), 256, 0, stream>>>(R0, wh3, cb3, g3, be3, m3, v3, R1);
  conv_bn_relu_wmma<128, 256, 16, false><<<dim3(4096, 4), 256, 0, stream>>>(R1, wh4, cb4, g4, be4, m4, v4, R0);
  maxpool2_kernel<256, 16><<<131072, 256, 0, stream>>>(R0, R1);
  gap_kernel<<<2048, 256, 0, stream>>>(R1, feats, featsh);

  // heads
  head_gemm_wmma<NE * NCLS, 0, false><<<dim3(16, 3), 256, 0, stream>>>(featsh, clsw, clsb, logits_e);
  head_gemm_wmma<NE * 128, 1, true ><<<dim3(16, 32), 256, 0, stream>>>(featsh, gw1, gb1, hidden);
  entropy_kernel<<<128, 256, 0, stream>>>(logits_e, conf);
  scores_kernel<<<128, 256, 0, stream>>>(hidden, gw2, gb2, conf, usage, routing);

  // routing + output
  route_kernel<<<1, 1024, 0, stream>>>(routing, Dout);
  final_kernel<<<8, 256, 0, stream>>>(routing, Dout, logits_e, out);
}
